// PerformerAttention_45552423141627
// MI455X (gfx1250) — compile-verified, matrix operand order checked by reading, not run on hardware
//
#include <hip/hip_runtime.h>
#include <hip/hip_bf16.h>
#include <hip/hip_fp16.h>

// ---- problem constants ----
#define DIMN    1024
#define HEADS   16
#define HD      64
#define NBF     266
#define NBF_PAD 288          // 18 tiles of 16 (padded features forced to 0)
#define BATCH   2
#define SEQ     4096
#define BH      (BATCH * HEADS)   // 32
#define BS      (BATCH * SEQ)     // 8192

typedef __attribute__((ext_vector_type(16))) _Float16 v16h;
typedef __attribute__((ext_vector_type(8)))  float    v8f;
typedef __attribute__((ext_vector_type(4)))  int      v4i;

// ---- CDNA5 async global->LDS staging (guarded; fallback = sync LDS copy) ----
#if defined(__has_builtin)
#  if __has_builtin(__builtin_amdgcn_global_load_async_to_lds_b128) && \
      __has_builtin(__builtin_amdgcn_s_wait_asynccnt)
#    define HAVE_ASYNC_LDS 1
#  endif
#endif
#ifndef HAVE_ASYNC_LDS
#  define HAVE_ASYNC_LDS 0
#endif

typedef __attribute__((address_space(1))) v4i GV4;   // global b128 payload
typedef __attribute__((address_space(3))) v4i LV4;   // LDS b128 payload

__device__ __forceinline__ void block_stage_lds(_Float16* lds_dst,
                                                const _Float16* gsrc,
                                                int nbytes) {
#if HAVE_ASYNC_LDS
  const char* s = (const char*)gsrc;
  char* d = (char*)lds_dst;
  for (int o = threadIdx.x * 16; o < nbytes; o += blockDim.x * 16) {
    __builtin_amdgcn_global_load_async_to_lds_b128(
        (GV4*)(unsigned long long)(s + o),
        (LV4*)(unsigned int)(unsigned long long)(d + o), 0, 0);
  }
  __builtin_amdgcn_s_wait_asynccnt(0);   // our own async loads complete
  __syncthreads();                       // everyone's loads visible
#else
  const int4* s4 = (const int4*)gsrc;
  int4* d4 = (int4*)lds_dst;
  const int n16 = nbytes >> 4;
  for (int i = threadIdx.x; i < n16; i += blockDim.x) d4[i] = s4[i];
  __syncthreads();
#endif
}

__device__ __forceinline__ v8f wmma32(v16h a, v16h b, v8f c) {
  // D = A(16x32 f16) * B(32x16 f16) + C(16x16 f32)
  return __builtin_amdgcn_wmma_f32_16x16x32_f16(false, a, false, b, (short)0, c,
                                                false, false);
}

// Load one 16x32 f16 operand tile from row-major memory (K contiguous).
// Works for global or LDS (generic) pointers -> global_load_b128 / ds_load_b128.
__device__ __forceinline__ v16h load_tile(const _Float16* base, int pitch) {
  int l = threadIdx.x & 31;
  return *(const v16h*)(base + (size_t)(l & 15) * pitch + ((l >> 4) << 4));
}

// ---------------- prep kernels ----------------
__global__ void cvt_f32_to_f16(const float* __restrict__ src,
                               _Float16* __restrict__ dst, int n) {
  for (int i = blockIdx.x * blockDim.x + threadIdx.x; i < n;
       i += gridDim.x * blockDim.x)
    dst[i] = (_Float16)src[i];
}

__global__ void prep_rf(const float* __restrict__ rf, _Float16* __restrict__ rfh) {
  int i = blockIdx.x * blockDim.x + threadIdx.x;
  if (i < NBF_PAD * HD) {
    int r = i / HD, c = i % HD;
    rfh[i] = (r < NBF) ? (_Float16)rf[r * HD + c] : (_Float16)0.0f;
  }
}

__global__ void zero_f32(float* __restrict__ p, int n) {
  int i = blockIdx.x * blockDim.x + threadIdx.x;
  if (i < n) p[i] = 0.0f;
}

// ---------------- QKV projection: C = X @ W^T + b ----------------
// grid: (BS/64, DIMN/128, 3); block 256 (8 waves).
// Wave = 16(M) x 64(N) strip: 4 accumulators, A reused across 4 WMMAs.
__global__ void qkv_gemm(const _Float16* __restrict__ Xh,
                         const _Float16* __restrict__ Wh,   // [3][DIMN][DIMN]
                         const float* __restrict__ qb, const float* __restrict__ kb,
                         const float* __restrict__ vb,
                         const unsigned char* __restrict__ mask,
                         _Float16* __restrict__ qh,   // [BH][SEQ][HD]
                         _Float16* __restrict__ kh,   // [BH][SEQ][HD]
                         _Float16* __restrict__ vT) { // [BH][HD][SEQ]
  const int z    = blockIdx.z;
  const int wave = threadIdx.x >> 5, lane = threadIdx.x & 31;
  const int m0 = blockIdx.x * 64 + (wave >> 1) * 16;
  const int n0 = blockIdx.y * 128 + (wave & 1) * 64;
  const _Float16* W    = Wh + (size_t)z * DIMN * DIMN;
  const float*    bias = (z == 0) ? qb : ((z == 1) ? kb : vb);

  const _Float16* Ab = Xh + (size_t)m0 * DIMN;
  const _Float16* Bb = W + (size_t)n0 * DIMN;
  v8f c[4] = {};
  for (int k = 0; k < DIMN; k += 32) {
    __builtin_prefetch(Ab + k + 256, 0, 1);
    __builtin_prefetch(Bb + k + 256, 0, 1);
    v16h a = load_tile(Ab + k, DIMN);
#pragma unroll
    for (int nt = 0; nt < 4; ++nt)
      c[nt] = wmma32(a, load_tile(Bb + (size_t)nt * 16 * DIMN + k, DIMN), c[nt]);
  }
  const int n = lane & 15, mb = (lane >> 4) << 3;
#pragma unroll
  for (int nt = 0; nt < 4; ++nt) {
    const int col = n0 + nt * 16 + n;
    const float bv = bias[col];
    const int h = col >> 6, hd = col & (HD - 1);
#pragma unroll
    for (int r = 0; r < 8; ++r) {
      int row = m0 + mb + r;
      int b_ = row >> 12, s = row & (SEQ - 1);
      size_t bh = (size_t)b_ * HEADS + h;
      float v = c[nt][r] + bv;
      if (z == 0) {
        qh[(bh * SEQ + s) * HD + hd] = (_Float16)v;
      } else if (z == 1) {
        kh[(bh * SEQ + s) * HD + hd] = (_Float16)v;
      } else {
        if (mask[b_ * SEQ + s]) v = 0.0f;     // zero masked v rows
        vT[(bh * HD + hd) * SEQ + s] = (_Float16)v;
      }
    }
  }
}

// ---------------- Performer feature map ----------------
// one wave per (bh, 16-row s tile); random features staged in LDS (async).
__global__ void feat_kernel(const _Float16* __restrict__ Xhm, // [BH][SEQ][HD]
                            const _Float16* __restrict__ rfh, // [NBF_PAD][HD]
                            const unsigned char* __restrict__ mask,
                            _Float16* __restrict__ outS,  // [BH][SEQ][NBF_PAD]
                            _Float16* __restrict__ outT,  // [BH][NBF_PAD][SEQ]
                            float* __restrict__ ksum,     // [BH][NBF_PAD]
                            int mode) {
  __shared__ _Float16 lrf[NBF_PAD * HD];                  // 36 KB
  block_stage_lds(lrf, rfh, NBF_PAD * HD * 2);

  const int wv   = blockIdx.x * (blockDim.x >> 5) + (threadIdx.x >> 5);
  const int lane = threadIdx.x & 31;
  const int bh = wv >> 8;            // SEQ/16 = 256 tiles per (b,h)
  const int s0 = (wv & 255) << 4;
  if (bh >= BH) return;

  const _Float16* x = Xhm + ((size_t)bh * SEQ + s0) * HD;
  v16h a0 = load_tile(x, HD);
  v16h a1 = load_tile(x + 32, HD);

  // per-row sum of squares in lanes 0..15
  float sq = 0.0f;
  if (lane < 16) {
    const _Float16* xr = x + (size_t)lane * HD;
    for (int d = 0; d < HD; ++d) { float t = (float)xr[d]; sq += t * t; }
  }
  float mrow = 0.0f;
  if (mode == 1 && lane < 16) {
    int b_ = bh >> 4;  // / HEADS
    mrow = mask[b_ * SEQ + s0 + lane] ? 1.0f : 0.0f;
  }

  const float norm   = 0.35355339059327373f;  // 64^-0.25
  const float hnorm2 = 0.5f * norm * norm;    // 0.0625
  const float ratio  = 0.06131393394849658f;  // 1/sqrt(266)

  for (int t = 0; t < NBF_PAD / 16; ++t) {
    const int f0 = t << 4;
    const _Float16* rb = lrf + (size_t)f0 * HD;           // B from LDS
    v8f c = {};
    c = wmma32(a0, load_tile(rb, HD), c);
    c = wmma32(a1, load_tile(rb + 32, HD), c);

    const int n = lane & 15, mb = (lane >> 4) << 3;
    const int feat = f0 + n;
    float acc = 0.0f;
#pragma unroll
    for (int r = 0; r < 8; ++r) {
      int m = mb + r;
      float sqm = __shfl(sq, m);                  // ds_bpermute broadcast
      float val = ratio * __expf(norm * c[r] - hnorm2 * sqm);
      if (feat >= NBF) val = 0.0f;                // zero padded features
      if (mode == 1) {
        if (__shfl(mrow, m) != 0.0f) val = 0.0f;  // key padding mask
        outT[((size_t)bh * NBF_PAD + feat) * SEQ + s0 + m] = (_Float16)val;
        acc += val;
      } else {
        outS[((size_t)bh * SEQ + s0 + m) * NBF_PAD + feat] = (_Float16)val;
      }
    }
    if (mode == 1) atomicAdd(&ksum[bh * NBF_PAD + feat], acc);
  }
}

// ---------------- kv = kf^T @ v  (K = SEQ = 4096) ----------------
// wave = 16 features x 64(all e): 4 accumulators, A reused across 4 WMMAs.
__global__ void kv_gemm(const _Float16* __restrict__ kfT, // [BH][NBF_PAD][SEQ]
                        const _Float16* __restrict__ vT,  // [BH][HD][SEQ]
                        _Float16* __restrict__ kvT) {     // [BH][HD][NBF_PAD]
  const int wv   = blockIdx.x * (blockDim.x >> 5) + (threadIdx.x >> 5);
  const int lane = threadIdx.x & 31;
  const int ft = wv % (NBF_PAD / 16);
  const int bh = wv / (NBF_PAD / 16);
  if (bh >= BH) return;

  const _Float16* A  = kfT + ((size_t)bh * NBF_PAD + ft * 16) * SEQ;
  const _Float16* Bb = vT + (size_t)bh * HD * SEQ;
  v8f c[4] = {};
  for (int k = 0; k < SEQ; k += 32) {
    __builtin_prefetch(A + k + 512, 0, 1);
    v16h a = load_tile(A + k, SEQ);
#pragma unroll
    for (int et = 0; et < 4; ++et)
      c[et] = wmma32(a, load_tile(Bb + (size_t)et * 16 * SEQ + k, SEQ), c[et]);
  }
  const int n = lane & 15, mb = (lane >> 4) << 3;
#pragma unroll
  for (int et = 0; et < 4; ++et) {
    const int e = et * 16 + n;
#pragma unroll
    for (int r = 0; r < 8; ++r) {
      int feat = ft * 16 + mb + r;
      kvT[((size_t)bh * HD + e) * NBF_PAD + feat] = (_Float16)c[et][r];
    }
  }
}

// ---------------- out = (qf @ kv) / (qf @ ksum + eps) ----------------
// 32 blocks per (b,h); per-(b,h) kv (36 KB) staged in LDS via async copy.
// A tiles cached in registers (K = 288).
__global__ void attn_out(const _Float16* __restrict__ qf,   // [BH][SEQ][NBF_PAD]
                         const _Float16* __restrict__ kvT,  // [BH][HD][NBF_PAD]
                         const float* __restrict__ ksum,    // [BH][NBF_PAD]
                         _Float16* __restrict__ attn) {     // [BATCH][SEQ][DIMN]
  const int bh   = blockIdx.x >> 5;            // uniform per block
  const int wave = threadIdx.x >> 5, lane = threadIdx.x & 31;
  const int s0 = ((((blockIdx.x & 31) << 3) + wave) << 4);

  __shared__ _Float16 lkv[HD * NBF_PAD];       // 36 KB
  block_stage_lds(lkv, kvT + (size_t)bh * HD * NBF_PAD, HD * NBF_PAD * 2);

  const _Float16* A = qf + ((size_t)bh * SEQ + s0) * NBF_PAD;
  v16h areg[NBF_PAD / 32];
#pragma unroll
  for (int i = 0; i < NBF_PAD / 32; ++i) areg[i] = load_tile(A + i * 32, NBF_PAD);

  // normalizer per row in lanes 0..15
  float nrm = 0.0f;
  if (lane < 16) {
    const _Float16* qr = A + (size_t)lane * NBF_PAD;
    const float* ks = ksum + bh * NBF_PAD;
    for (int m = 0; m < NBF_PAD; ++m) nrm += (float)qr[m] * ks[m];
  }

  const int b_ = bh >> 4, h = bh & (HEADS - 1);
  for (int et = 0; et < HD / 16; ++et) {
    const _Float16* Bm = lkv + (size_t)et * 16 * NBF_PAD;   // B from LDS
    v8f c = {};
#pragma unroll
    for (int i = 0; i < NBF_PAD / 32; ++i)
      c = wmma32(areg[i], load_tile(Bm + i * 32, NBF_PAD), c);

    const int n = lane & 15, mb = (lane >> 4) << 3;
#pragma unroll
    for (int r = 0; r < 8; ++r) {
      int m = mb + r;
      float nv = __shfl(nrm, m) + 1e-6f;
      attn[((size_t)b_ * SEQ + s0 + m) * DIMN + h * HD + et * 16 + n] =
          (_Float16)(c[r] / nv);
    }
  }
}

// ---------------- final projection: out = attn @ o_w^T + o_b (f32) ----------------
__global__ void out_proj(const _Float16* __restrict__ attn,
                         const _Float16* __restrict__ Woh,
                         const float* __restrict__ ob, float* __restrict__ out) {
  const int wave = threadIdx.x >> 5, lane = threadIdx.x & 31;
  const int m0 = blockIdx.x * 64 + (wave >> 1) * 16;
  const int n0 = blockIdx.y * 128 + (wave & 1) * 64;
  const _Float16* Ab = attn + (size_t)m0 * DIMN;
  const _Float16* Bb = Woh + (size_t)n0 * DIMN;
  v8f c[4] = {};
  for (int k = 0; k < DIMN; k += 32) {
    __builtin_prefetch(Ab + k + 256, 0, 1);
    __builtin_prefetch(Bb + k + 256, 0, 1);
    v16h a = load_tile(Ab + k, DIMN);
#pragma unroll
    for (int nt = 0; nt < 4; ++nt)
      c[nt] = wmma32(a, load_tile(Bb + (size_t)nt * 16 * DIMN + k, DIMN), c[nt]);
  }
  const int n = lane & 15, mb = (lane >> 4) << 3;
#pragma unroll
  for (int nt = 0; nt < 4; ++nt) {
    const float bv = ob[n0 + nt * 16 + n];
#pragma unroll
    for (int r = 0; r < 8; ++r)
      out[(size_t)(m0 + mb + r) * DIMN + n0 + nt * 16 + n] = c[nt][r] + bv;
  }
}

// ---------------- host launcher ----------------
extern "C" void kernel_launch(void* const* d_in, const int* in_sizes, int n_in,
                              void* d_out, int out_size, void* d_ws, size_t ws_size,
                              hipStream_t stream) {
  const float* query = (const float*)d_in[0];
  const float* q_w   = (const float*)d_in[1];
  const float* q_b   = (const float*)d_in[2];
  const float* k_w   = (const float*)d_in[3];
  const float* k_b   = (const float*)d_in[4];
  const float* v_w   = (const float*)d_in[5];
  const float* v_b   = (const float*)d_in[6];
  const float* o_w   = (const float*)d_in[7];
  const float* o_b   = (const float*)d_in[8];
  const float* rf    = (const float*)d_in[9];
  const unsigned char* mask = (const unsigned char*)d_in[10];  // jax bool -> 1B

  char* ws = (char*)d_ws;
  size_t off = 0;
  auto take = [&](size_t bytes) -> void* {
    void* p = ws + off;
    off += (bytes + 255) & ~(size_t)255;
    return p;
  };
  _Float16* Xh   = (_Float16*)take((size_t)BS * DIMN * 2);
  _Float16* Wh   = (_Float16*)take((size_t)4 * DIMN * DIMN * 2);  // q,k,v,o
  _Float16* rfh  = (_Float16*)take((size_t)NBF_PAD * HD * 2);
  _Float16* qh   = (_Float16*)take((size_t)BH * SEQ * HD * 2);
  _Float16* kh   = (_Float16*)take((size_t)BH * SEQ * HD * 2);
  _Float16* vT   = (_Float16*)take((size_t)BH * HD * SEQ * 2);
  _Float16* qf   = (_Float16*)take((size_t)BH * SEQ * NBF_PAD * 2);
  _Float16* kfT  = (_Float16*)take((size_t)BH * NBF_PAD * SEQ * 2);
  float*    ksum = (float*)take((size_t)BH * NBF_PAD * 4);
  _Float16* kvT  = (_Float16*)take((size_t)BH * HD * NBF_PAD * 2);
  _Float16* attn = (_Float16*)take((size_t)BS * DIMN * 2);
  (void)ws_size; (void)n_in; (void)in_sizes; (void)out_size;

  // stage f16 operands
  cvt_f32_to_f16<<<2048, 256, 0, stream>>>(query, Xh, BS * DIMN);
  cvt_f32_to_f16<<<1024, 256, 0, stream>>>(q_w, Wh + (size_t)0 * DIMN * DIMN, DIMN * DIMN);
  cvt_f32_to_f16<<<1024, 256, 0, stream>>>(k_w, Wh + (size_t)1 * DIMN * DIMN, DIMN * DIMN);
  cvt_f32_to_f16<<<1024, 256, 0, stream>>>(v_w, Wh + (size_t)2 * DIMN * DIMN, DIMN * DIMN);
  cvt_f32_to_f16<<<1024, 256, 0, stream>>>(o_w, Wh + (size_t)3 * DIMN * DIMN, DIMN * DIMN);
  prep_rf<<<(NBF_PAD * HD + 255) / 256, 256, 0, stream>>>(rf, rfh);
  zero_f32<<<(BH * NBF_PAD + 255) / 256, 256, 0, stream>>>(ksum, BH * NBF_PAD);

  // QKV projection (register-blocked WMMA GEMM)
  dim3 gB(BS / 64, DIMN / 128, 3);
  qkv_gemm<<<gB, 256, 0, stream>>>(Xh, Wh, q_b, k_b, v_b, mask, qh, kh, vT);

  // feature maps (rf staged in LDS via async copy)
  const int featBlocks = (BH * (SEQ / 16)) / 8;  // 1024
  feat_kernel<<<featBlocks, 256, 0, stream>>>(qh, rfh, mask, qf, nullptr, nullptr, 0);
  feat_kernel<<<featBlocks, 256, 0, stream>>>(kh, rfh, mask, nullptr, kfT, ksum, 1);

  // kv accumulation
  const int kvBlocks = (BH * (NBF_PAD / 16)) / 8;  // 72
  kv_gemm<<<kvBlocks, 256, 0, stream>>>(kfT, vT, kvT);

  // numerator / normalizer (kv staged in LDS via async copy)
  attn_out<<<featBlocks, 256, 0, stream>>>(qf, kvT, ksum, attn);

  // output projection
  dim3 gO(BS / 64, DIMN / 128);
  out_proj<<<gO, 256, 0, stream>>>(attn, Wh + (size_t)3 * DIMN * DIMN, o_b,
                                   (float*)d_out);
}
